// GuidedAttention_53395033424139
// MI455X (gfx1250) — matrix operand order repaired and checked by reading, not compiled, not run on hardware
//
#include <hip/hip_runtime.h>
#include <hip/hip_bf16.h>

typedef __attribute__((ext_vector_type(16))) _Float16 v16h;
typedef __attribute__((ext_vector_type(8)))  float    v8f;

constexpr int Bsz = 16, C = 64, O = 16, HID = 32, H = 128, W = 128;
constexpr int HW = H * W;                 // 16384
constexpr int NPIX = Bsz * HW;            // 262144
constexpr int NTILES = NPIX / 16;         // 16384 (16 pixels per WMMA tile)

// Load a 16x32 f16 A-matrix tile from a row-major [rows, ncols] f32 weight
// matrix in the contiguous-k A order used here (B tiles match):
//   lanes 0-15 : row M = m*16 + lane,     K = kt*32 + {0..15}
//   lanes 16-31: row M = m*16 + lane-16,  K = kt*32 + {16..31}
// VGPR j holds packed (K=2j, K=2j+1) halves.
__device__ inline v16h load_wmma_a_tile(const float* __restrict__ Wmat, int ncols,
                                        int m, int kt, int lane) {
    const int hi  = lane >> 4;
    const int row = m * 16 + (lane & 15);
    const float* p = Wmat + row * ncols + kt * 32 + hi * 16;
    v16h a;
#pragma unroll
    for (int j = 0; j < 4; ++j) {                   // VGPRs 0..3: K = base+0..7
        a[2 * j]     = (_Float16)p[2 * j];
        a[2 * j + 1] = (_Float16)p[2 * j + 1];
    }
    const float* q = p + 8;                         // VGPRs 4..7: K = base+8..15
#pragma unroll
    for (int j = 0; j < 4; ++j) {
        a[8 + 2 * j] = (_Float16)q[2 * j];
        a[9 + 2 * j] = (_Float16)q[2 * j + 1];
    }
    return a;
}

// Load a 32(ch) x 16(px) f16 B-matrix tile from activations laid out [.., k, pix]
// with channel stride HW. `src` is pre-offset to (batch, pixel) so index = k*HW.
// Contiguous-k B order (matches A): lanes 0-15 hold K=0..15, lanes 16-31 hold K=16..31,
// column N = lane&15, VGPR j packs (K=2j, K=2j+1). Streamed (non-temporal):
// every element is consumed exactly once kernel-wide.
__device__ inline v16h load_wmma_b_tile(const float* __restrict__ src, int kt,
                                        int lane) {
    const int kbase = kt * 32 + (lane >> 4) * 16;
    v16h bv;
#pragma unroll
    for (int j = 0; j < 8; ++j) {
        float x = __builtin_nontemporal_load(src + (size_t)(kbase + 2 * j) * HW);
        float y = __builtin_nontemporal_load(src + (size_t)(kbase + 2 * j + 1) * HW);
        bv[2 * j]     = (_Float16)x;
        bv[2 * j + 1] = (_Float16)y;
    }
    return bv;
}

// tanh(x) = copysign(1 - 2/(e^{2|x|}+1), x); v_exp/v_rcp TRANS ops co-issue
// with VALU. e^inf -> rcp=0 -> +/-1, so saturation is handled for free.
__device__ inline float fast_tanh(float x) {
    float e = __expf(2.0f * fabsf(x));
    float r = __builtin_fmaf(-2.0f, __builtin_amdgcn_rcpf(e + 1.0f), 1.0f);
    return copysignf(r, x);
}

#define WMMA_F16(A, B, Cacc) \
    __builtin_amdgcn_wmma_f32_16x16x32_f16(false, (A), false, (B), (short)0, (Cacc), false, false)

__global__ void __launch_bounds__(256)
guided_attention_kernel(const float* __restrict__ input, const float* __restrict__ guide,
                        const float* __restrict__ Wi, const float* __restrict__ bi,
                        const float* __restrict__ Wg, const float* __restrict__ bg,
                        const float* __restrict__ Wa, const float* __restrict__ ba,
                        float* __restrict__ out) {
    const int lane = threadIdx.x & 31;
    const int hi   = lane >> 4;
    const int n    = lane & 15;
    const int wave   = blockIdx.x * (blockDim.x >> 5) + (threadIdx.x >> 5);
    const int nwaves = gridDim.x * (blockDim.x >> 5);

    // ---- Stage all weights in registers (tiny: 8KB+8KB+2KB, L2-resident) ----
    v16h ai[2][2], ag[2][2];
#pragma unroll
    for (int m = 0; m < 2; ++m)
#pragma unroll
        for (int kt = 0; kt < 2; ++kt) {
            ai[m][kt] = load_wmma_a_tile(Wi, C, m, kt, lane);
            ag[m][kt] = load_wmma_a_tile(Wg, C, m, kt, lane);
        }
    v16h aa = load_wmma_a_tile(Wa, HID, 0, 0, lane);

    // All-ones A tile: channel-sum of input as a WMMA (row m of D = sum_k B[k,n]).
    v16h ones;
#pragma unroll
    for (int j = 0; j < 16; ++j) ones[j] = (_Float16)1.0f;

    // Bias folded into WMMA C accumulator seeds.
    // C/D f32 layout: VGPR r -> M = r + (lane>=16 ? 8 : 0), N = lane&15.
    v8f cbias[2], cba;
#pragma unroll
    for (int m = 0; m < 2; ++m)
#pragma unroll
        for (int r = 0; r < 8; ++r) {
            int idx = m * 16 + hi * 8 + r;
            cbias[m][r] = bi[idx] + bg[idx];
        }
#pragma unroll
    for (int r = 0; r < 8; ++r) cba[r] = ba[hi * 8 + r];

    for (int tile = wave; tile < NTILES; tile += nwaves) {
        const int pixbase = tile * 16;          // 16 consecutive pixels, same (b,row)
        const int b   = pixbase >> 14;          // / HW
        const int off = pixbase & (HW - 1);
        const float* inp = input + (size_t)b * C * HW + off + n;
        const float* gui = guide + (size_t)b * C * HW + off + n;

        v16h bin0 = load_wmma_b_tile(inp, 0, lane);
        v16h bin1 = load_wmma_b_tile(inp, 1, lane);
        v16h bgu0 = load_wmma_b_tile(gui, 0, lane);
        v16h bgu1 = load_wmma_b_tile(gui, 1, lane);

        // hidden = Wi*x + bi + Wg*g + bg  (HID=32 -> two 16-row M tiles)
        v8f acc0 = cbias[0], acc1 = cbias[1];
        acc0 = WMMA_F16(ai[0][0], bin0, acc0);
        acc0 = WMMA_F16(ai[0][1], bin1, acc0);
        acc0 = WMMA_F16(ag[0][0], bgu0, acc0);
        acc0 = WMMA_F16(ag[0][1], bgu1, acc0);
        acc1 = WMMA_F16(ai[1][0], bin0, acc1);
        acc1 = WMMA_F16(ai[1][1], bin1, acc1);
        acc1 = WMMA_F16(ag[1][0], bgu0, acc1);
        acc1 = WMMA_F16(ag[1][1], bgu1, acc1);

        // channel sum of input per pixel via ones-GEMM (every row of D is the sum)
        v8f accS = {};
        accS = WMMA_F16(ones, bin0, accS);
        accS = WMMA_F16(ones, bin1, accS);
        const float s = accS[0];

        // relu, then reshape D layout (M striped across VGPRs) into B layout
        // (K striped across lanes) for the attention GEMM via one cross-half swap.
        float vals[16];
#pragma unroll
        for (int r = 0; r < 8; ++r) {
            float a0 = fmaxf(acc0[r], 0.0f);
            float a1 = fmaxf(acc1[r], 0.0f);
            float o0 = __shfl_xor(a0, 16, 32);  // partner half's k =  8+r (lo) /    r (hi)
            float o1 = __shfl_xor(a1, 16, 32);  // partner half's k = 24+r (lo) / 16+r (hi)
            vals[r]     = hi ? o1 : a0;         // k = base + r
            vals[8 + r] = hi ? a1 : o0;         // k = base + 8 + r
        }
        v16h hB;
#pragma unroll
        for (int j = 0; j < 8; ++j) {
            hB[2 * j]     = (_Float16)vals[2 * j];
            hB[2 * j + 1] = (_Float16)vals[2 * j + 1];
        }

        // attention = tanh(Wa*h + ba); out = attention * sum_c(input)
        v8f att = WMMA_F16(aa, hB, cba);

        float* outp = out + (size_t)b * O * HW + off + n;
#pragma unroll
        for (int r = 0; r < 8; ++r) {
            int o = hi * 8 + r;
            __builtin_nontemporal_store(fast_tanh(att[r]) * s, outp + (size_t)o * HW);
        }
    }
}

extern "C" void kernel_launch(void* const* d_in, const int* in_sizes, int n_in,
                              void* d_out, int out_size, void* d_ws, size_t ws_size,
                              hipStream_t stream) {
    (void)in_sizes; (void)n_in; (void)out_size; (void)d_ws; (void)ws_size;
    const float* input = (const float*)d_in[0];
    const float* guide = (const float*)d_in[1];
    const float* Wi    = (const float*)d_in[2];
    const float* bi    = (const float*)d_in[3];
    const float* Wg    = (const float*)d_in[4];
    const float* bg    = (const float*)d_in[5];
    const float* Wa    = (const float*)d_in[6];
    const float* ba    = (const float*)d_in[7];
    float* out = (float*)d_out;

    // 16384 tiles / (512 blocks * 8 waves) = 4 tiles per wave:
    // amortizes weight staging 2x better than before while still leaving
    // 4096 waves in flight to hide HBM latency at 23.3 TB/s.
    guided_attention_kernel<<<512, 256, 0, stream>>>(input, guide, Wi, bi, Wg, bg,
                                                     Wa, ba, out);
}
